// RBN_14542759264629
// MI455X (gfx1250) — matrix operand order repaired and checked by reading, not compiled, not run on hardware
//
#include <hip/hip_runtime.h>
#include <hip/hip_bf16.h>

// ---------------------------------------------------------------------------
// Fused RBF network forward for MI455X (gfx1250, wave32, WMMA).
//   out = exp(-BETA * (|x|^2 + |c|^2 - 2 x.cT)) @ W + b
// Fusion avoids the 512MB round-trip of the [B,C] rbf intermediate (~22us at
// 23.3 TB/s, dwarfing ~15us of bf16-WMMA math). This revision double-buffers
// the per-chunk centers/W staging with CDNA5 async global->LDS DMA
// (GLOBAL_LOAD_ASYNC_TO_LDS_B128, ASYNCcnt) so staging latency fully overlaps
// the WMMA chain of the previous chunk.
// ---------------------------------------------------------------------------

typedef __attribute__((ext_vector_type(16))) __bf16 v16bf;
typedef __attribute__((ext_vector_type(8)))  float  v8f;

#define B_ROWS 16384
#define D_DIM  512
#define C_DIM  4096
#define O_DIM  128
#define BETA_F 0.35f

#define BM 64   // x rows per workgroup
#define CB 32   // centers per chunk (== bf16 WMMA K)

// LDS layout (dynamic LDS base == offset 0: kernel has no static __shared__)
#define OFF_XA 0                               // [BM][D] bf16      64 KB
#define OFF_CB (BM * D_DIM * 2)                // 2 x [CB][D] bf16  64 KB
#define OFF_WT (OFF_CB + 2 * CB * D_DIM * 2)   // 2 x [O][CB] bf16  16 KB
#define OFF_RB (OFF_WT + 2 * O_DIM * CB * 2)   // [BM][CB] bf16      4 KB
#define OFF_XQ (OFF_RB + BM * CB * 2)          // [BM] f32         256 B
#define SMEM_BYTES (OFF_XQ + BM * 4)           // 151,808 B

// round-to-nearest-even float -> bf16 (as raw ushort), no __bf16 arithmetic
__device__ __forceinline__ unsigned short f2bf(float f) {
    unsigned int u = __builtin_bit_cast(unsigned int, f);
    unsigned int r = u + 0x7FFFu + ((u >> 16) & 1u);
    return (unsigned short)(r >> 16);
}

// async DMA: 16 bytes per lane, global -> LDS, tracked by ASYNCcnt
__device__ __forceinline__ void async_cp16(unsigned lds_off, const void* gaddr) {
    asm volatile("global_load_async_to_lds_b128 %0, %1, off"
                 :: "v"(lds_off), "v"(gaddr)
                 : "memory");
}

// ---------------- prep kernels (run once per launch, tiny cost) -------------

// centers [C,D] f32 -> bf16, plus row norms |c|^2
__global__ void prep_centers(const float* __restrict__ cen,
                             unsigned short* __restrict__ cenB,
                             float* __restrict__ c_sq) {
    int row = blockIdx.x;
    int t   = threadIdx.x;  // 128 threads
    const float4* src = (const float4*)(cen + (size_t)row * D_DIM);
    float s = 0.f;
    for (int i = t; i < D_DIM / 4; i += 128) {
        float4 v = src[i];
        s += v.x * v.x + v.y * v.y + v.z * v.z + v.w * v.w;
        unsigned short* d = cenB + (size_t)row * D_DIM + i * 4;
        d[0] = f2bf(v.x); d[1] = f2bf(v.y); d[2] = f2bf(v.z); d[3] = f2bf(v.w);
    }
    __shared__ float red[128];
    red[t] = s; __syncthreads();
    for (int off = 64; off > 0; off >>= 1) {
        if (t < off) red[t] += red[t + off];
        __syncthreads();
    }
    if (t == 0) c_sq[row] = red[0];
}

// x row norms |x|^2
__global__ void prep_xsq(const float* __restrict__ x, float* __restrict__ x_sq) {
    int row = blockIdx.x;
    int t   = threadIdx.x;  // 128 threads
    const float4* src = (const float4*)(x + (size_t)row * D_DIM);
    float s = 0.f;
    for (int i = t; i < D_DIM / 4; i += 128) {
        float4 v = src[i];
        s += v.x * v.x + v.y * v.y + v.z * v.z + v.w * v.w;
    }
    __shared__ float red[128];
    red[t] = s; __syncthreads();
    for (int off = 64; off > 0; off >>= 1) {
        if (t < off) red[t] += red[t + off];
        __syncthreads();
    }
    if (t == 0) x_sq[row] = red[0];
}

// W [C,O] f32 -> Wt [O,C] bf16 (transposed so per-lane B-operand is contiguous)
__global__ void prep_w(const float* __restrict__ W, unsigned short* __restrict__ Wt) {
    int idx = blockIdx.x * blockDim.x + threadIdx.x;
    if (idx >= C_DIM * O_DIM) return;
    int c = idx / O_DIM, o = idx % O_DIM;
    Wt[(size_t)o * C_DIM + c] = f2bf(W[idx]);
}

// ---------------- main fused kernel ----------------------------------------
// 256 threads = 8 waves. Wave w: mw = w&3 (16-row group), nh = w>>2.
// Cross GEMM: wave computes 16x16 tile (rows mw*16.., centers nh*16..).
// Second GEMM: wave owns rows mw*16.., output cols nh*64 .. nh*64+63.

// issue one chunk's async copies into LDS buffer `bufsel`
// (10 async instructions per wave: 8 for centers, 2 for W)
__device__ __forceinline__ void issue_chunk(const unsigned short* __restrict__ cenB,
                                            const unsigned short* __restrict__ Wt,
                                            int cc, unsigned bufsel, int tid) {
    // centers chunk: CB*D bf16 = 32KB = 2048 x 16B vectors, WG-wide
    unsigned cb_off = OFF_CB + bufsel * (CB * D_DIM * 2);
    const char* csrc = (const char*)(cenB + (size_t)cc * D_DIM);
#pragma unroll
    for (int k = 0; k < 8; k++) {
        int i = tid + k * 256;                       // 16B vector index
        async_cp16(cb_off + i * 16, csrc + (size_t)i * 16);
    }
    // W chunk: wTs[o][k] = Wt[o][cc+k]; O*CB bf16 = 8KB = 512 x 16B vectors
    unsigned wt_off = OFF_WT + bufsel * (O_DIM * CB * 2);
#pragma unroll
    for (int k = 0; k < 2; k++) {
        int i = tid + k * 256;
        int o = i >> 2, q = i & 3;
        async_cp16(wt_off + i * 16, (const char*)(Wt + (size_t)o * C_DIM + cc + q * 8));
    }
}

__launch_bounds__(256)
__global__ void rbf_fused(const float* __restrict__ x,
                          const unsigned short* __restrict__ cenB,
                          const unsigned short* __restrict__ Wt,
                          const float* __restrict__ c_sq,
                          const float* __restrict__ x_sq,
                          const float* __restrict__ bias,
                          float* __restrict__ out) {
    extern __shared__ char smem[];
    unsigned short* xA  = (unsigned short*)(smem + OFF_XA);
    unsigned short* rbs = (unsigned short*)(smem + OFF_RB);
    float*          xqs = (float*)(smem + OFF_XQ);

    const int tid   = threadIdx.x;
    const int wave  = tid >> 5;
    const int lane  = tid & 31;
    const int lhalf = lane >> 4;   // which 16-lane half
    const int l16   = lane & 15;
    const int mw    = wave & 3;
    const int nh    = wave >> 2;
    const int m0    = blockIdx.x * BM;

    // stage this block's x tile as bf16 (single pass over x, no reuse elsewhere)
    {
        const float4* src = (const float4*)(x + (size_t)m0 * D_DIM);
        for (int i = tid; i < BM * D_DIM / 4; i += 256) {
            float4 v = src[i];
            unsigned short* d = xA + i * 4;
            d[0] = f2bf(v.x); d[1] = f2bf(v.y); d[2] = f2bf(v.z); d[3] = f2bf(v.w);
        }
        if (tid < BM) xqs[tid] = x_sq[m0 + tid];
    }

    // kick off DMA for chunk 0 into buffer 0
    issue_chunk(cenB, Wt, 0, 0, tid);

    v8f oacc[4];
    for (int n2 = 0; n2 < 4; n2++)
        for (int k = 0; k < 8; k++) oacc[n2][k] = 0.f;

    const int NCHUNK = C_DIM / CB;
    for (int it = 0; it < NCHUNK; ++it) {
        const int cc = it * CB;
        const unsigned cur = it & 1u;
        const unsigned short* cBs = (const unsigned short*)(smem + OFF_CB + cur * (CB * D_DIM * 2));
        const unsigned short* wTs = (const unsigned short*)(smem + OFF_WT + cur * (O_DIM * CB * 2));

        __syncthreads();  // all waves done reading buf[cur^1] (gemm2 of it-1) and rbs

        if (it + 1 < NCHUNK) {
            issue_chunk(cenB, Wt, cc + CB, cur ^ 1u, tid);   // overlap DMA with compute
            if (it + 2 < NCHUNK)                             // warm L2 one chunk further
                __builtin_prefetch(cenB + (size_t)(cc + 2 * CB) * D_DIM + tid * 64, 0, 0);
            asm volatile("s_wait_asynccnt 0xa" ::: "memory");  // chunk `it` landed
        } else {
            asm volatile("s_wait_asynccnt 0x0" ::: "memory");  // drain: last chunk landed
        }
        __syncthreads();  // chunk `it` visible workgroup-wide

        // ---- cross GEMM: 16x16 tile over K = D = 512 ----
        v8f acc;
        for (int k = 0; k < 8; k++) acc[k] = 0.f;
        const unsigned short* arow = xA  + (size_t)(mw * 16 + l16) * D_DIM;
        const unsigned short* brow = cBs + (size_t)(nh * 16 + l16) * D_DIM;
        for (int kk = 0; kk < D_DIM; kk += 32) {
            v16bf a, bmat;
            // A layout: lanes<16 hold K 0..7 & 16..23, lanes>=16 hold K 8..15 & 24..31
            ((uint4*)&a)[0] = *(const uint4*)(arow + kk + lhalf * 8);
            ((uint4*)&a)[1] = *(const uint4*)(arow + kk + 16 + lhalf * 8);
            // B layout: lane's column = l16, same K split as A (8h..8h+7 & 16+8h..)
            ((uint4*)&bmat)[0] = *(const uint4*)(brow + kk + lhalf * 8);
            ((uint4*)&bmat)[1] = *(const uint4*)(brow + kk + 16 + lhalf * 8);
            acc = __builtin_amdgcn_wmma_f32_16x16x32_bf16(
                false, a, false, bmat, (short)0, acc, false, false);
        }

        // ---- rbf: distances + exp, write A-tile for second GEMM to LDS ----
        float cq = c_sq[cc + nh * 16 + l16];
        for (int v = 0; v < 8; v++) {
            int mloc = mw * 16 + v + lhalf * 8;     // C/D layout: VGPR v, lane half
            float sq = xqs[mloc] + cq - 2.0f * acc[v];
            float r  = __expf(-BETA_F * sq);        // v_exp_f32
            rbs[mloc * CB + nh * 16 + l16] = f2bf(r);
        }
        __syncthreads();

        // ---- second GEMM: rbf[16x32] @ W[32x64] accumulate ----
        {
            v16bf ra;
            const unsigned short* rrow = rbs + (size_t)(mw * 16 + l16) * CB;
            ((uint4*)&ra)[0] = *(const uint4*)(rrow + lhalf * 8);
            ((uint4*)&ra)[1] = *(const uint4*)(rrow + 16 + lhalf * 8);
            for (int n2 = 0; n2 < 4; n2++) {
                int o = nh * 64 + n2 * 16 + l16;
                v16bf wb;
                ((uint4*)&wb)[0] = *(const uint4*)(wTs + o * CB + lhalf * 8);
                ((uint4*)&wb)[1] = *(const uint4*)(wTs + o * CB + 16 + lhalf * 8);
                oacc[n2] = __builtin_amdgcn_wmma_f32_16x16x32_bf16(
                    false, ra, false, wb, (short)0, oacc[n2], false, false);
            }
        }
    }

    // epilogue: add bias, store f32 output
    for (int n2 = 0; n2 < 4; n2++) {
        int o = nh * 64 + n2 * 16 + l16;
        float bo = bias[o];
        for (int v = 0; v < 8; v++) {
            int m = m0 + mw * 16 + v + lhalf * 8;
            out[(size_t)m * O_DIM + o] = oacc[n2][v] + bo;
        }
    }
}

// ---------------------------------------------------------------------------

extern "C" void kernel_launch(void* const* d_in, const int* in_sizes, int n_in,
                              void* d_out, int out_size, void* d_ws, size_t ws_size,
                              hipStream_t stream) {
    const float* x    = (const float*)d_in[0];
    const float* cen  = (const float*)d_in[1];
    const float* W    = (const float*)d_in[2];
    const float* bias = (const float*)d_in[3];
    float* out        = (float*)d_out;

    // workspace carve-up (~5.3 MB total)
    char* ws = (char*)d_ws;
    unsigned short* cenB = (unsigned short*)ws;                                // 4 MB
    unsigned short* Wt   = (unsigned short*)(ws + (size_t)C_DIM * D_DIM * 2);  // 1 MB
    float* c_sq = (float*)(ws + (size_t)C_DIM * D_DIM * 2
                              + (size_t)C_DIM * O_DIM * 2);                    // 16 KB
    float* x_sq = (float*)((char*)c_sq + (size_t)C_DIM * 4);                   // 64 KB

    prep_centers<<<C_DIM, 128, 0, stream>>>(cen, cenB, c_sq);
    prep_xsq<<<B_ROWS, 128, 0, stream>>>(x, x_sq);
    prep_w<<<(C_DIM * O_DIM + 255) / 256, 256, 0, stream>>>(W, Wt);

    rbf_fused<<<B_ROWS / BM, 256, SMEM_BYTES, stream>>>(x, cenB, Wt, c_sq, x_sq, bias, out);
}
